// Fingerprint_38173669326857
// MI455X (gfx1250) — matrix operand order repaired and checked by reading, not compiled, not run on hardware
//
#include <hip/hip_runtime.h>
#include <hip/hip_bf16.h>
#include <math.h>

#define Bsz  256
#define Nn   128
#define Kn   6
#define FEAT 39
#define BOND 10
#define CAT  49          // FEAT + BOND
#define FPd  200
#define RR   3
#define TT   2
#define NEGC (-900000000.0f)

#define BN   (Bsz * Nn)          // 32768
#define BNK  (Bsz * Nn * Kn)     // 196608

typedef __attribute__((ext_vector_type(16))) _Float16 v16h;
typedef __attribute__((ext_vector_type(8)))  _Float16 v8h;
typedef __attribute__((ext_vector_type(8)))  float    v8f;

__device__ __forceinline__ float leakyf(float x)  { return x >= 0.f ? x : 0.01f * x; }
__device__ __forceinline__ float sigmoidf_(float x){ return 1.f / (1.f + __expf(-x)); }
__device__ __forceinline__ float eluf(float x)    { return x > 0.f ? x : __expf(x) - 1.f; }

// ---------------------------------------------------------------------------
// f32 -> f16 staging with zero padding: dst[rowsP x colsP], src[rows x cols]
// ---------------------------------------------------------------------------
__global__ void cvt_pad(const float* __restrict__ src, _Float16* __restrict__ dst,
                        int rows, int cols, int rowsP, int colsP)
{
    int idx = blockIdx.x * blockDim.x + threadIdx.x;
    if (idx >= rowsP * colsP) return;
    int c = idx % colsP, r = idx / colsP;
    float v = (r < rows && c < cols) ? src[(size_t)r * cols + c] : 0.f;
    dst[(size_t)r * colsP + c] = (_Float16)v;
}

// ---------------------------------------------------------------------------
// WMMA GEMM on padded f16 operands:
//   out[M,Nout] = ACT( Ah[Mp,Kp] * Wh[NP2,Kp]^T + bias ) (+ res)
// Ah rows padded to 16, Wh rows padded to 32, Kp multiple of 32 (zero fill).
// One 16x32 output strip per wave: unpredicated b128 loads, 2 WMMA per K-step.
// ACT: 0 = bias only, 1 = leaky, 2 = leaky then + residual
// ---------------------------------------------------------------------------
template<int ACT>
__global__ void __launch_bounds__(128)
wmma_gemm(const _Float16* __restrict__ Ah, const _Float16* __restrict__ Wh,
          const float* __restrict__ bias, const float* __restrict__ res,
          float* __restrict__ out, int M, int Nout, int Kp)
{
    const int lane    = threadIdx.x & 31;
    const int wave    = threadIdx.x >> 5;
    const int tilesN2 = (Nout + 31) >> 5;                // 32-wide N strips
    const int tile    = blockIdx.x * 4 + wave;
    const int tM      = tile / tilesN2;
    const int tN2     = tile - tM * tilesN2;
    if (tM * 16 >= M) return;                            // wave-uniform exit

    const int l16 = lane & 15;
    const int hi  = lane >> 4;                           // lane group 0/1
    const _Float16* Arow = Ah + (size_t)(tM * 16 + l16) * Kp;   // A: M = lane%16
    const _Float16* W0   = Wh + (size_t)(tN2 * 32 + l16) * Kp;  // B: N = lane%16
    const _Float16* W1   = W0 + (size_t)16 * Kp;

    v8f c0 = {}, c1 = {};
    for (int k0 = 0; k0 < Kp; k0 += 32) {
        // A 16x32 f16 layout: lanes 0-15 K=k0..k0+15, lanes 16-31 K=+16
        v8h x0 = *(const v8h*)(Arow + k0 + hi * 16);
        v8h x1 = *(const v8h*)(Arow + k0 + 8 + hi * 16);
        v16h a = __builtin_shufflevector(x0, x1, 0,1,2,3,4,5,6,7,8,9,10,11,12,13,14,15);
        // B 32x16 f16 layout: lanes 0-15 K=k0..k0+15, lanes 16-31 K=+16
        v8h y0 = *(const v8h*)(W0 + k0 + hi * 16);
        v8h y1 = *(const v8h*)(W0 + k0 + 8 + hi * 16);
        v16h b0 = __builtin_shufflevector(y0, y1, 0,1,2,3,4,5,6,7,8,9,10,11,12,13,14,15);
        v8h z0 = *(const v8h*)(W1 + k0 + hi * 16);
        v8h z1 = *(const v8h*)(W1 + k0 + 8 + hi * 16);
        v16h b1 = __builtin_shufflevector(z0, z1, 0,1,2,3,4,5,6,7,8,9,10,11,12,13,14,15);
        c0 = __builtin_amdgcn_wmma_f32_16x16x32_f16(false, a, false, b0,
                                                    (short)0, c0, false, false);
        c1 = __builtin_amdgcn_wmma_f32_16x16x32_f16(false, a, false, b1,
                                                    (short)0, c1, false, false);
    }

    const int col0 = tN2 * 32 + l16;
    const int col1 = col0 + 16;
#pragma unroll
    for (int r = 0; r < 8; ++r) {
        int row = tM * 16 + r + hi * 8;                  // C/D: M = r + 8*(lane>=16)
        if (row >= M) continue;
        if (col0 < Nout) {
            float v = c0[r] + bias[col0];
            if (ACT >= 1) v = leakyf(v);
            if (ACT == 2) v += res[(size_t)row * Nout + col0];
            out[(size_t)row * Nout + col0] = v;
        }
        if (col1 < Nout) {
            float v = c1[r] + bias[col1];
            if (ACT >= 1) v = leakyf(v);
            if (ACT == 2) v += res[(size_t)row * Nout + col1];
            out[(size_t)row * Nout + col1] = v;
        }
    }
}

// out[r] = dot(X[r, 0:cols], w) + (*addp)
__global__ void __launch_bounds__(128)
rowdot(const float* __restrict__ X, const float* __restrict__ w,
       const float* __restrict__ addp, float* __restrict__ out,
       int rows, int cols)
{
    int row  = blockIdx.x * 4 + (threadIdx.x >> 5);
    int lane = threadIdx.x & 31;
    if (row >= rows) return;
    const float* x = X + (size_t)row * cols;
    float s = 0.f;
    for (int c = lane; c < cols; c += 32) s += x[c] * w[c];
#pragma unroll
    for (int off = 16; off > 0; off >>= 1) s += __shfl_down(s, off, 32);
    if (lane == 0) out[row] = s + (addp ? addp[0] : 0.f);
}

// Acat[bnk, 0:49] = concat(atom_list[b, adl[bnk]], bond_list[b, bdl[bnk]])
__global__ void gather_concat(const float* __restrict__ atom,
                              const float* __restrict__ bond,
                              const int* __restrict__ adl,
                              const int* __restrict__ bdl,
                              float* __restrict__ Acat)
{
    int idx = blockIdx.x * blockDim.x + threadIdx.x;
    if (idx >= BNK * CAT) return;
    int c   = idx % CAT;
    int bnk = idx / CAT;
    int b   = bnk / (Nn * Kn);
    float v;
    if (c < FEAT) {
        int j = adl[bnk];
        v = atom[((size_t)(b * Nn + j)) * FEAT + c];
    } else {
        int j = bdl[bnk];
        v = bond[((size_t)(b * Nn + j)) * BOND + (c - FEAT)];
    }
    Acat[(size_t)bnk * CAT + c] = v;
}

// Round-1 attention combine: per (b,n) softmax over K=6, ctx = elu(sum w*nftK)
__global__ void __launch_bounds__(64)
attn_combine_r1(const float* __restrict__ sL, const float* __restrict__ sRk,
                const int* __restrict__ adl, const float* __restrict__ nftK,
                float* __restrict__ ctx)
{
    int bn = blockIdx.x;
    float s[Kn], wgt[Kn];
    float mx = -3.4e38f;
#pragma unroll
    for (int k = 0; k < Kn; ++k) {
        int j = adl[bn * Kn + k];
        float sc = leakyf(sL[bn] + sRk[bn * Kn + k]);
        if (j == Nn - 1) sc += NEGC;
        s[k] = sc; mx = fmaxf(mx, sc);
    }
    float sum = 0.f;
#pragma unroll
    for (int k = 0; k < Kn; ++k) { wgt[k] = __expf(s[k] - mx); sum += wgt[k]; }
    float inv = 1.f / sum;
#pragma unroll
    for (int k = 0; k < Kn; ++k)
        wgt[k] = (adl[bn * Kn + k] == Nn - 1) ? 0.f : wgt[k] * inv;
    for (int f = threadIdx.x; f < FPd; f += 64) {
        float acc = 0.f;
#pragma unroll
        for (int k = 0; k < Kn; ++k)
            acc += wgt[k] * nftK[((size_t)bn * Kn + k) * FPd + f];
        ctx[(size_t)bn * FPd + f] = eluf(acc);
    }
}

// Rounds 2-3: scores from per-node dots, nft gathered from per-node transform
__global__ void __launch_bounds__(64)
attn_combine_gather(const float* __restrict__ sL, const float* __restrict__ sRn,
                    const int* __restrict__ adl, const float* __restrict__ nftN,
                    float* __restrict__ ctx)
{
    int bn = blockIdx.x;
    int b  = bn / Nn;
    int jj[Kn]; float s[Kn], wgt[Kn];
    float mx = -3.4e38f;
#pragma unroll
    for (int k = 0; k < Kn; ++k) {
        int j = adl[bn * Kn + k]; jj[k] = j;
        float sc = leakyf(sL[bn] + sRn[b * Nn + j]);
        if (j == Nn - 1) sc += NEGC;
        s[k] = sc; mx = fmaxf(mx, sc);
    }
    float sum = 0.f;
#pragma unroll
    for (int k = 0; k < Kn; ++k) { wgt[k] = __expf(s[k] - mx); sum += wgt[k]; }
    float inv = 1.f / sum;
#pragma unroll
    for (int k = 0; k < Kn; ++k) wgt[k] = (jj[k] == Nn - 1) ? 0.f : wgt[k] * inv;
    for (int f = threadIdx.x; f < FPd; f += 64) {
        float acc = 0.f;
#pragma unroll
        for (int k = 0; k < Kn; ++k)
            acc += wgt[k] * nftN[((size_t)b * Nn + jj[k]) * FPd + f];
        ctx[(size_t)bn * FPd + f] = eluf(acc);
    }
}

// GRU nonlinear combine (torch GRUCell, gates r,z,n); optional leaky out
__global__ void gru_combine(const float* __restrict__ gi, const float* __restrict__ gh,
                            const float* __restrict__ h, float* __restrict__ hout,
                            float* __restrict__ actout, int rows)
{
    int idx = blockIdx.x * blockDim.x + threadIdx.x;
    if (idx >= rows * FPd) return;
    int f = idx % FPd, r_ = idx / FPd;
    const float* gir = gi + (size_t)r_ * 3 * FPd;
    const float* ghr = gh + (size_t)r_ * 3 * FPd;
    float rr = sigmoidf_(gir[f]           + ghr[f]);
    float zz = sigmoidf_(gir[FPd + f]     + ghr[FPd + f]);
    float nn = tanhf(    gir[2 * FPd + f] + rr * ghr[2 * FPd + f]);
    float hp = h[(size_t)r_ * FPd + f];
    float hn = (1.f - zz) * nn + zz * hp;
    hout[(size_t)r_ * FPd + f] = hn;
    if (actout) actout[(size_t)r_ * FPd + f] = leakyf(hn);
}

__global__ void molsum(const float* __restrict__ act, const float* __restrict__ mask,
                       float* __restrict__ molf, float* __restrict__ actmol)
{
    int idx = blockIdx.x * blockDim.x + threadIdx.x;
    if (idx >= Bsz * FPd) return;
    int f = idx % FPd, b = idx / FPd;
    float s = 0.f;
    for (int n = 0; n < Nn; ++n)
        s += act[((size_t)b * Nn + n) * FPd + f] * mask[b * Nn + n];
    molf[idx] = s; actmol[idx] = leakyf(s);
}

// per-molecule softmax over N=128 atoms + weighted ELU combine
__global__ void __launch_bounds__(128)
mol_combine(const float* __restrict__ sMol, const float* __restrict__ sAtom,
            const float* __restrict__ mask, const float* __restrict__ aftN,
            float* __restrict__ molctx)
{
    __shared__ float wsh[Nn];
    __shared__ float red[Nn];
    int b = blockIdx.x, t = threadIdx.x;
    float m  = mask[b * Nn + t];
    float sc = leakyf(sMol[b] + sAtom[b * Nn + t]);
    if (m == 0.f) sc += NEGC;
    red[t] = sc; __syncthreads();
    for (int o = 64; o > 0; o >>= 1) { if (t < o) red[t] = fmaxf(red[t], red[t + o]); __syncthreads(); }
    float mx = red[0]; __syncthreads();
    float e = __expf(sc - mx);
    red[t] = e; __syncthreads();
    for (int o = 64; o > 0; o >>= 1) { if (t < o) red[t] += red[t + o]; __syncthreads(); }
    float inv = 1.f / red[0]; __syncthreads();
    wsh[t] = e * inv * m; __syncthreads();
    for (int f = t; f < FPd; f += Nn) {
        float acc = 0.f;
        for (int n = 0; n < Nn; ++n)
            acc += wsh[n] * aftN[((size_t)b * Nn + n) * FPd + f];
        molctx[(size_t)b * FPd + f] = eluf(acc);
    }
}

__global__ void __launch_bounds__(32)
final_out(const float* __restrict__ r2, const float* __restrict__ out_w,
          const float* __restrict__ out_b, const float* __restrict__ sw,
          const float* __restrict__ sb, float* __restrict__ pred)
{
    int b = blockIdx.x, lane = threadIdx.x;
    float s = 0.f;
    for (int c = lane; c < FPd; c += 32) s += r2[(size_t)b * FPd + c] * out_w[c];
#pragma unroll
    for (int off = 16; off > 0; off >>= 1) s += __shfl_down(s, off, 32);
    if (lane == 0) pred[b] = (s + out_b[0]) * sw[0] + sb[0];
}

// ---------------------------------------------------------------------------
struct GemmStage { _Float16* Ah; _Float16* Wh; };

static inline void run_gemm(int actmode, const float* A, const float* W,
                            const float* bias, const float* res, float* out,
                            int M, int Nout, int Ka, GemmStage st, hipStream_t s)
{
    int Mp  = (M + 15) & ~15;
    int Kp  = (Ka + 31) & ~31;
    int NP2 = (Nout + 31) & ~31;
    {
        int tot = Mp * Kp;
        cvt_pad<<<(tot + 255) / 256, 256, 0, s>>>(A, st.Ah, M, Ka, Mp, Kp);
        tot = NP2 * Kp;
        cvt_pad<<<(tot + 255) / 256, 256, 0, s>>>(W, st.Wh, Nout, Ka, NP2, Kp);
    }
    int tiles  = (Mp / 16) * (NP2 / 32);
    int blocks = (tiles + 3) / 4;
    if (actmode == 0)      wmma_gemm<0><<<blocks, 128, 0, s>>>(st.Ah, st.Wh, bias, res, out, M, Nout, Kp);
    else if (actmode == 1) wmma_gemm<1><<<blocks, 128, 0, s>>>(st.Ah, st.Wh, bias, res, out, M, Nout, Kp);
    else                   wmma_gemm<2><<<blocks, 128, 0, s>>>(st.Ah, st.Wh, bias, res, out, M, Nout, Kp);
}

extern "C" void kernel_launch(void* const* d_in, const int* in_sizes, int n_in,
                              void* d_out, int out_size, void* d_ws, size_t ws_size,
                              hipStream_t stream)
{
    (void)in_sizes; (void)n_in; (void)out_size; (void)ws_size;
    const float* atom_list   = (const float*)d_in[0];
    const float* bond_list   = (const float*)d_in[1];
    const float* atom_mask   = (const float*)d_in[2];
    const float* atom_fc_w   = (const float*)d_in[3];
    const float* atom_fc_b   = (const float*)d_in[4];
    const float* nb_fc_w     = (const float*)d_in[5];
    const float* nb_fc_b     = (const float*)d_in[6];
    const float* align_w     = (const float*)d_in[7];   // [R,1,400]
    const float* align_b     = (const float*)d_in[8];   // [R,1]
    const float* attend_w    = (const float*)d_in[9];   // [R,200,200]
    const float* attend_b    = (const float*)d_in[10];  // [R,200]
    const float* gru_wih     = (const float*)d_in[11];  // [R,600,200]
    const float* gru_whh     = (const float*)d_in[12];
    const float* gru_bih     = (const float*)d_in[13];  // [R,600]
    const float* gru_bhh     = (const float*)d_in[14];
    const float* mol_align_w = (const float*)d_in[15];  // [1,400]
    const float* mol_align_b = (const float*)d_in[16];
    const float* mol_att_w   = (const float*)d_in[17];  // [200,200]
    const float* mol_att_b   = (const float*)d_in[18];
    const float* mgru_wih    = (const float*)d_in[19];
    const float* mgru_whh    = (const float*)d_in[20];
    const float* mgru_bih    = (const float*)d_in[21];
    const float* mgru_bhh    = (const float*)d_in[22];
    const float* lin1_w      = (const float*)d_in[23];
    const float* lin1_b      = (const float*)d_in[24];
    const float* lin2_w      = (const float*)d_in[25];
    const float* lin2_b      = (const float*)d_in[26];
    const float* out_w       = (const float*)d_in[27];
    const float* out_b       = (const float*)d_in[28];
    const float* scaler_w    = (const float*)d_in[29];
    const float* scaler_b    = (const float*)d_in[30];
    const int*   adl         = (const int*)d_in[31];
    const int*   bdl         = (const int*)d_in[32];

    float* hbuf = (float*)d_out;                         // atom_feature / GRU h
    float* pred = hbuf + (size_t)BN * FPd;

    float* ws = (float*)d_ws;
    size_t off = 0;
    auto alloc = [&](size_t nf) { float* p = ws + off; off += (nf + 63) & ~(size_t)63; return p; };
    float* neigh  = alloc((size_t)BNK * FPd);            // [BNK,200]
    float* slab   = alloc((size_t)BNK * FPd);            // Acat -> nftK -> gi||gh
    float* act    = alloc((size_t)BN * FPd);
    float* ctx    = alloc((size_t)BN * FPd);
    float* nftN   = alloc((size_t)BN * FPd);             // also aftN
    float* sL     = alloc(BN);
    float* sR     = alloc(BNK);                          // round1 per-edge; later per-node
    float* molf   = alloc((size_t)Bsz * FPd);
    float* actmol = alloc((size_t)Bsz * FPd);
    float* molctx = alloc((size_t)Bsz * FPd);
    float* sAtom  = alloc(BN);
    float* sMol   = alloc(Bsz);
    float* mgi    = alloc((size_t)Bsz * 3 * FPd);
    float* mgh    = alloc((size_t)Bsz * 3 * FPd);
    float* r1b    = alloc((size_t)Bsz * FPd);
    float* r2b    = alloc((size_t)Bsz * FPd);
    // f16 staging: max A = BNK x 224 halves, max W = 608x224 (round to 640x224)
    GemmStage st;
    st.Ah = (_Float16*)alloc(((size_t)BNK * 224) / 2);
    st.Wh = (_Float16*)alloc(((size_t)640 * 224) / 2);

    float* Acat = slab;                 // [BNK,49]
    float* nftK = slab;                 // [BNK,200]
    float* gi   = slab;                 // [BN,600]
    float* gh   = slab + (size_t)BN * 3 * FPd;

    // atom_feature = leaky(atom_list @ atom_fc_w.T + b) -> hbuf (also h0)
    run_gemm(1, atom_list, atom_fc_w, atom_fc_b, nullptr, hbuf, BN, FPd, FEAT, st, stream);

    // neighbor_feature = leaky(concat(gathered) @ nb_fc_w.T + b)
    {
        int tot = BNK * CAT;
        gather_concat<<<(tot + 255) / 256, 256, 0, stream>>>(atom_list, bond_list, adl, bdl, Acat);
        run_gemm(1, Acat, nb_fc_w, nb_fc_b, nullptr, neigh, BNK, FPd, CAT, st, stream);
    }

    for (int d = 0; d < RR; ++d) {
        const float* awL = align_w + (size_t)d * 2 * FPd;
        const float* awR = awL + FPd;
        const float* src = (d == 0) ? hbuf : act;        // af for attention
        rowdot<<<(BN + 3) / 4, 128, 0, stream>>>(src, awL, align_b + d, sL, BN, FPd);
        if (d == 0) {
            rowdot<<<(BNK + 3) / 4, 128, 0, stream>>>(neigh, awR, nullptr, sR, BNK, FPd);
            run_gemm(0, neigh, attend_w, attend_b, nullptr, nftK, BNK, FPd, FPd, st, stream);
            attn_combine_r1<<<BN, 64, 0, stream>>>(sL, sR, adl, nftK, ctx);
        } else {
            rowdot<<<(BN + 3) / 4, 128, 0, stream>>>(act, awR, nullptr, sR, BN, FPd);
            run_gemm(0, act, attend_w + (size_t)d * FPd * FPd, attend_b + (size_t)d * FPd,
                     nullptr, nftN, BN, FPd, FPd, st, stream);
            attn_combine_gather<<<BN, 64, 0, stream>>>(sL, sR, adl, nftN, ctx);
        }
        // GRU gates via WMMA, then nonlinear combine (h lives in d_out)
        run_gemm(0, ctx,  gru_wih + (size_t)d * 3 * FPd * FPd, gru_bih + (size_t)d * 3 * FPd,
                 nullptr, gi, BN, 3 * FPd, FPd, st, stream);
        run_gemm(0, hbuf, gru_whh + (size_t)d * 3 * FPd * FPd, gru_bhh + (size_t)d * 3 * FPd,
                 nullptr, gh, BN, 3 * FPd, FPd, st, stream);
        gru_combine<<<(BN * FPd + 255) / 256, 256, 0, stream>>>(gi, gh, hbuf, hbuf, act, BN);
    }

    // ---- molecule phase ----
    molsum<<<(Bsz * FPd + 255) / 256, 256, 0, stream>>>(act, atom_mask, molf, actmol);
    // loop-invariant: aft = act @ mol_attend_w.T + b ; sAtom = act . wR
    run_gemm(0, act, mol_att_w, mol_att_b, nullptr, nftN, BN, FPd, FPd, st, stream);
    rowdot<<<(BN + 3) / 4, 128, 0, stream>>>(act, mol_align_w + FPd, nullptr, sAtom, BN, FPd);

    for (int t = 0; t < TT; ++t) {
        rowdot<<<(Bsz + 3) / 4, 128, 0, stream>>>(actmol, mol_align_w, mol_align_b, sMol, Bsz, FPd);
        mol_combine<<<Bsz, Nn, 0, stream>>>(sMol, sAtom, atom_mask, nftN, molctx);
        run_gemm(0, molctx, mgru_wih, mgru_bih, nullptr, mgi, Bsz, 3 * FPd, FPd, st, stream);
        run_gemm(0, molf,   mgru_whh, mgru_bhh, nullptr, mgh, Bsz, 3 * FPd, FPd, st, stream);
        gru_combine<<<(Bsz * FPd + 255) / 256, 256, 0, stream>>>(mgi, mgh, molf, molf, actmol, Bsz);
    }

    // head: r1 = leaky(mol@lin1)+mol ; r2 = leaky(r1@lin2)+r1 ; pred
    run_gemm(2, molf, lin1_w, lin1_b, molf, r1b, Bsz, FPd, FPd, st, stream);
    run_gemm(2, r1b,  lin2_w, lin2_b, r1b,  r2b, Bsz, FPd, FPd, st, stream);
    final_out<<<Bsz, 32, 0, stream>>>(r2b, out_w, out_b, scaler_w, scaler_b, pred);
}
